// Slice_231928234078
// MI455X (gfx1250) — compile-verified, matrix operand order checked by reading, not run on hardware
//
#include <hip/hip_runtime.h>

typedef __attribute__((ext_vector_type(16))) _Float16 v16h;
typedef __attribute__((ext_vector_type(8)))  float    v8f;
typedef __attribute__((ext_vector_type(4)))  float    v4f;

namespace {

constexpr int kN = 4, kC = 12, kD = 8, kHg = 16, kWg = 16, kH = 1024, kW = 1024;
constexpr int kNumWaves = 8;

__device__ __forceinline__ int imin(int a, int b) { return a < b ? a : b; }
__device__ __forceinline__ int imax(int a, int b) { return a > b ? a : b; }

// One workgroup owns the pixel rectangle of cell band (xb, yb) of batch n.
// Inside it every trilerp corner lies in the 2x2x8 = 32 texel window
// {xb,xb+1} x {yb,yb+1} x {0..7}, so out[pix,c] = A(16x32) * B(32x16)
// maps exactly onto v_wmma_f32_16x16x32_f16 with B invariant per wave.
__global__ __launch_bounds__(256) void slice_wmma_kernel(
    const float* __restrict__ grid,    // [N, C, D, Hg, Wg]
    const float* __restrict__ guide,   // [N, 1, H, W]
    float* __restrict__ out)           // [N, C, H, W]
{
  const int bid = blockIdx.x;
  const int yb = bid & 15;            // y-cell band (columns)
  const int xb = (bid >> 4) & 15;     // x-cell band (rows)
  const int n  = bid >> 8;

  const int lane = threadIdx.x & 31;
  const int wave = threadIdx.x >> 5;
  const int half = lane >> 4;         // 0: lanes 0-15, 1: lanes 16-31
  const int m    = lane & 15;

  // Rows h with floor(h*15/1023)==xb:  [ceil(1023*xb/15), ceil(1023*(xb+1)/15))
  const int hs = (1023 * xb + 14) / 15;
  const int he = imin(kH, (1023 * (xb + 1) + 14) / 15);
  const int ws = (1023 * yb + 14) / 15;
  const int we = imin(kW, (1023 * (yb + 1) + 14) / 15);
  const int nRows   = he - hs;
  const int wlo     = ws & ~15;                       // 16-aligned chunk base
  const int nChunks = (we - wlo + 15) >> 4;
  const int items   = nRows * nChunks;

  // ---- Build B (32x16, f16): B[k][c] = grid[n][c][z][yy][xx], k = z*4+j*2+i.
  // 16-bit B layout: lane = column (channel); lanes 0-15 carry K = 0-7,16-23
  // in vector slots e=0..15; lanes 16-31 carry K = 8-15,24-31.
  const int cc = m;                   // channel this lane feeds (cols 12-15 = 0)
  v16h B = {};
  if (cc < kC) {
    const int kb = half ? 8 : 0;
#pragma unroll
    for (int e = 0; e < 16; ++e) {
      const int K = (e < 8) ? (kb + e) : (kb + 8 + e);  // kb+16+(e-8)
      const int z = K >> 2, j = (K >> 1) & 1, i = K & 1;
      const int yy = imin(yb + j, kHg - 1);             // weight is 0 if clamped
      const int xx = imin(xb + i, kWg - 1);
      B[e] = (_Float16)grid[(((n * kC + cc) * kD + z) * kHg + yy) * kWg + xx];
    }
  }

  const float inv = 15.0f / 1023.0f;
  const int zb = half ? 2 : 0;        // this lane's z slots: zb, zb+1, zb+4, zb+5

  // ---- Software-pipelined item loop: issue guide load for item i+1 before
  // the WMMA of item i. All pipeline control flow is wave-uniform, so the
  // WMMA always executes with full EXEC.
  int  item = wave;
  bool live = item < items;
  int  h = hs, w0 = wlo, w = ws;
  float g = 0.0f;
  if (live) {
    h  = hs + (item % nRows);
    w0 = wlo + (item / nRows) * 16;
    w  = imin(imax(w0 + m, ws), we - 1);
    g  = guide[(n * kH + h) * kW + w];
  }

  while (live) {
    // ---- Stage next iteration's indices + guide load (overlaps the WMMA).
    const int  itemN = item + kNumWaves;
    const bool liveN = itemN < items;
    int hN = h, w0N = w0, wN = w;
    float gN = 0.0f;
    if (liveN) {
      hN  = hs + (itemN % nRows);
      w0N = wlo + (itemN / nRows) * 16;
      wN  = imin(imax(w0N + m, ws), we - 1);
      gN  = guide[(n * kH + hN) * kW + wN];
    }

    // ---- Weights for the current item.
    const float wx = (float)h * inv - (float)xb;        // floor(ix)==xb in band
    const float wy = (float)w * inv - (float)yb;        // floor(iy)==yb in band
    const float iz = (g + 1.0f) * (0.5f * (kD - 1));
    const float fz = floorf(iz);
    const float wz = iz - fz;
    const int  z0  = imin(imax((int)fz, 0), kD - 1);

    const float q00 = (1.0f - wx) * (1.0f - wy);
    const float q10 = wx * (1.0f - wy);
    const float q01 = (1.0f - wx) * wy;
    const float q11 = wx * wy;

    // ---- Build A row for pixel m (split across lane m / lane m+16).
    v16h A = {};
#pragma unroll
    for (int t = 0; t < 4; ++t) {
      const int z = zb + (t & 1) + ((t >> 1) << 2);     // zb, zb+1, zb+4, zb+5
      const float zw = (z == z0) ? (1.0f - wz)
                                 : ((z == z0 + 1) ? wz : 0.0f);
      const int e = t * 4;                              // K = z*4 .. z*4+3
      A[e + 0] = (_Float16)(zw * q00);
      A[e + 1] = (_Float16)(zw * q10);
      A[e + 2] = (_Float16)(zw * q01);
      A[e + 3] = (_Float16)(zw * q11);
    }

    v8f acc = {};
    acc = __builtin_amdgcn_wmma_f32_16x16x32_f16(
        /*neg_a=*/false, A, /*neg_b=*/false, B,
        /*c_mod=*/(short)0, acc, /*reuse_a=*/false, /*reuse_b=*/false);

    // ---- Store. D layout: lane = channel, VGPR r = pixel (half -> r+8).
    if (cc < kC) {
      const long rowbase = ((long)(n * kC + cc) * kH + h) * kW;
      if (w0 >= ws && w0 + 16 <= we) {
        // Interior chunk: contiguous, 16B aligned -> two b128 stores.
        const long base = rowbase + w0 + half * 8;
        v4f lo = {acc[0], acc[1], acc[2], acc[3]};
        v4f hi = {acc[4], acc[5], acc[6], acc[7]};
        *(v4f*)(out + base)     = lo;
        *(v4f*)(out + base + 4) = hi;
      } else {
        // Ragged band edge: branchless dword stores to clamped addresses.
        // Clamped lanes computed with the same clamped w, so duplicate
        // writes carry bit-identical values (single owner wave per chunk).
#pragma unroll
        for (int p = 0; p < 8; ++p) {
          const int wp = imin(imax(w0 + half * 8 + p, ws), we - 1);
          out[rowbase + wp] = acc[p];
        }
      }
    }

    item = itemN; live = liveN;
    h = hN; w0 = w0N; w = wN; g = gN;
  }
}

}  // namespace

extern "C" void kernel_launch(void* const* d_in, const int* /*in_sizes*/,
                              int /*n_in*/, void* d_out, int /*out_size*/,
                              void* /*d_ws*/, size_t /*ws_size*/,
                              hipStream_t stream) {
  const float* grid  = (const float*)d_in[0];   // [4,12,8,16,16] f32
  const float* guide = (const float*)d_in[1];   // [4,1,1024,1024] f32
  float* out = (float*)d_out;                   // [4,12,1024,1024] f32

  dim3 blocks(kN * 16 * 16);                    // (n, x-band, y-band)
  slice_wmma_kernel<<<blocks, 256, 0, stream>>>(grid, guide, out);
}
